// SFANet_54958401519625
// MI455X (gfx1250) — matrix operand order repaired and checked by reading, not compiled, not run on hardware
//
#include <hip/hip_runtime.h>

// ---------------------------------------------------------------------------
// MI455X (gfx1250, wave32). All GEMM-shaped work -> V_WMMA_F32_16X16X32_F16.
// Each wave computes a 32x64 output strip (2 M-tiles x 4 N-tiles), K step 32:
// 8 WMMAs per 12 b128 loads. A-gathers are branch-free (clamp + select).
// ---------------------------------------------------------------------------

typedef _Float16 f16;
typedef __attribute__((ext_vector_type(16))) _Float16 v16h;
typedef __attribute__((ext_vector_type(8)))  _Float16 v8h;
typedef __attribute__((ext_vector_type(8)))  float    v8f;

#define F_BIAS   1
#define F_BN     2
#define F_RELU   4
#define F_RES    8
#define F_OUT_H 16
#define F_OUT_F 32

// BN_INV = 1/sqrt(1 + 1e-5)
#define BN_INV 0.99999500003749975f

// ---------------------------------------------------------------------------
// Pack weight matrix into WMMA B-fragment order.
// t = ((ntile*KB + kb)*32 + lane)*16 + j
//   lane holds column N = ntile*16 + (lane&15), K = kb*32 + 8*(lane>=16) + (j&7) + 16*(j>=8)
// mode 0: W row-major [N,K] (dense linears, 1x1 convs OIHW)
// mode 1: W OIHW 3x3; K ordered as (tap=(ky,kx), ci)
// ---------------------------------------------------------------------------
__global__ void pack_B_kernel(const float* __restrict__ W, f16* __restrict__ Bp,
                              int N, int K, int Cin, int mode)
{
    int KB = (K + 31) >> 5;
    long total = (long)(N >> 4) * KB * 512;
    long t = (long)blockIdx.x * blockDim.x + threadIdx.x;
    if (t >= total) return;
    int j  = (int)(t & 15);
    int L  = (int)((t >> 4) & 31);
    long blk = t >> 9;
    int kb = (int)(blk % KB);
    int ntile = (int)(blk / KB);
    int n = ntile * 16 + (L & 15);
    int k = kb * 32 + ((L >> 4) << 3) + (j & 7) + ((j & 8) << 1);
    float v = 0.0f;
    if (k < K) {
        if (mode == 0) {
            v = W[(long)n * K + k];
        } else {
            int tap = k / Cin;
            int ci  = k - tap * Cin;
            int ky = tap / 3, kx = tap - ky * 3;
            v = W[(((long)n * Cin + ci) * 3 + ky) * 3 + kx];
        }
    }
    Bp[t] = (f16)v;
}

// ---------------------------------------------------------------------------
// A-fragment loader (one 16x32 fragment row per lane), branch-free.
// AMODE 0: dense row-major [M,K], K%32==0
// AMODE 1: NHWC 1x1 conv w/ stride (vector if K%32==0, else clamped scalar)
// AMODE 2: NHWC 3x3 conv pad=1, Cin pow2 >= 8 (two contiguous 8-groups)
// AMODE 3: NHWC 3x3 conv pad=1, generic Cin (clamped scalar)
// ---------------------------------------------------------------------------
template <int AMODE>
__device__ __forceinline__ v16h load_afrag(
    const f16* __restrict__ A, long rowbase, int n_img, int ho, int wo,
    int k0, int K, int Hin, int Win, int Cin, int stride, int cinShift,
    bool kAligned)
{
    v16h a;
    if (AMODE == 0 || (AMODE == 1 && kAligned)) {
        v8h lo = *(const v8h*)(A + rowbase + k0);
        v8h hi = *(const v8h*)(A + rowbase + k0 + 16);
        #pragma unroll
        for (int j = 0; j < 8; ++j) { a[j] = lo[j]; a[8 + j] = hi[j]; }
    } else if (AMODE == 1) {
        #pragma unroll
        for (int j = 0; j < 16; ++j) {
            int k = k0 + (j & 7) + ((j & 8) << 1);
            int kc = k < K ? k : K - 1;
            f16 val = A[rowbase + kc];
            a[j] = (k < K) ? val : (f16)0.0f;
        }
    } else if (AMODE == 2) {
        #pragma unroll
        for (int g = 0; g < 2; ++g) {
            int kBase = k0 + g * 16;
            int tap = kBase >> cinShift;
            int ci  = kBase & (Cin - 1);
            int ky = tap / 3, kx = tap - ky * 3;
            int hi2 = ho * stride + ky - 1;
            int wi2 = wo * stride + kx - 1;
            bool inb = ((unsigned)hi2 < (unsigned)Hin) &&
                       ((unsigned)wi2 < (unsigned)Win);
            int hc = inb ? hi2 : 0;
            int wc = inb ? wi2 : 0;
            v8h d = *(const v8h*)(A + (((long)n_img * Hin + hc) * Win + wc) * Cin + ci);
            #pragma unroll
            for (int j = 0; j < 8; ++j) a[g * 8 + j] = inb ? d[j] : (f16)0.0f;
        }
    } else {
        #pragma unroll
        for (int j = 0; j < 16; ++j) {
            int k = k0 + (j & 7) + ((j & 8) << 1);
            int kc = k < K ? k : K - 1;
            int tap = kc / Cin;
            int ci  = kc - tap * Cin;
            int ky = tap / 3, kx = tap - ky * 3;
            int hi2 = ho * stride + ky - 1;
            int wi2 = wo * stride + kx - 1;
            bool inb = ((unsigned)hi2 < (unsigned)Hin) &&
                       ((unsigned)wi2 < (unsigned)Win) && (k < K);
            int hc = ((unsigned)hi2 < (unsigned)Hin) ? hi2 : 0;
            int wc = ((unsigned)wi2 < (unsigned)Win) ? wi2 : 0;
            f16 val = A[(((long)n_img * Hin + hc) * Win + wc) * Cin + ci];
            a[j] = inb ? val : (f16)0.0f;
        }
    }
    return a;
}

// ---------------------------------------------------------------------------
// WMMA GEMM: out[M,N] = A * B (+ fused epilogue). One wave -> 32x64 strip.
// Requires M%32==0, N%64==0. Wave-uniform control keeps EXEC all-1s.
// ---------------------------------------------------------------------------
template <int AMODE, int FLAGS>
__global__ __launch_bounds__(128) void gemm_wmma(
    const f16* __restrict__ A, const f16* __restrict__ Bp,
    const float* __restrict__ bias, const float* __restrict__ bng,
    const float* __restrict__ bnb, const f16* __restrict__ res,
    f16* __restrict__ outh, float* __restrict__ outf,
    int M, int N, int K,
    int Hout, int Wout, int Hin, int Win, int Cin, int stride,
    int cinShift, float scale)
{
    const int KB = (K + 31) >> 5;
    const int Ngrp = N >> 6;                            // 4 N-tiles per wave
    const int Mgrp = M >> 5;                            // 2 M-tiles per wave
    const int wave = blockIdx.x * (blockDim.x >> 5) + (threadIdx.x >> 5);
    if (wave >= Mgrp * Ngrp) return;                    // wave-uniform
    const int ng = wave % Ngrp;
    const int mg = wave / Ngrp;
    const int ntile0 = ng * 4;
    const int mtile0 = mg * 2;
    const int lane = threadIdx.x & 31;
    const int lhi = lane >> 4;                          // 0 or 1
    const int m0 = mtile0 * 16 + (lane & 15);
    const int m1 = m0 + 16;

    long rb0 = 0, rb1 = 0;
    int ni0 = 0, ho0 = 0, wo0 = 0, ni1 = 0, ho1 = 0, wo1 = 0;
    if (AMODE == 0) {
        rb0 = (long)m0 * K;
        rb1 = (long)m1 * K;
    } else {
        int hw = Hout * Wout;
        ni0 = m0 / hw; { int r = m0 - ni0 * hw; ho0 = r / Wout; wo0 = r - ho0 * Wout; }
        ni1 = m1 / hw; { int r = m1 - ni1 * hw; ho1 = r / Wout; wo1 = r - ho1 * Wout; }
        if (AMODE == 1) {
            rb0 = (((long)ni0 * Hin + ho0 * stride) * Win + wo0 * stride) * Cin;
            rb1 = (((long)ni1 * Hin + ho1 * stride) * Win + wo1 * stride) * Cin;
        }
    }

    const bool kAligned = (K & 31) == 0;
    v8f c00 = {}, c01 = {}, c02 = {}, c03 = {};
    v8f c10 = {}, c11 = {}, c12 = {}, c13 = {};
    const f16* bpBase = Bp + ((long)ntile0 * KB * 32 + lane) * 16;
    const long bTS = (long)KB * 512;                    // elements per N-tile

    for (int kb = 0; kb < KB; ++kb) {
        const int k0 = kb * 32 + lhi * 8;
        v16h a0 = load_afrag<AMODE>(A, rb0, ni0, ho0, wo0, k0, K, Hin, Win, Cin, stride, cinShift, kAligned);
        v16h a1 = load_afrag<AMODE>(A, rb1, ni1, ho1, wo1, k0, K, Hin, Win, Cin, stride, cinShift, kAligned);
        const f16* bp = bpBase + (long)kb * 512;
        {
            v8h blo = *(const v8h*)bp;
            v8h bhi = *(const v8h*)(bp + 8);
            v16h b;
            #pragma unroll
            for (int j = 0; j < 8; ++j) { b[j] = blo[j]; b[8 + j] = bhi[j]; }
            c00 = __builtin_amdgcn_wmma_f32_16x16x32_f16(false, a0, false, b, (short)0, c00, false, false);
            c10 = __builtin_amdgcn_wmma_f32_16x16x32_f16(false, a1, false, b, (short)0, c10, false, false);
        }
        {
            v8h blo = *(const v8h*)(bp + bTS);
            v8h bhi = *(const v8h*)(bp + bTS + 8);
            v16h b;
            #pragma unroll
            for (int j = 0; j < 8; ++j) { b[j] = blo[j]; b[8 + j] = bhi[j]; }
            c01 = __builtin_amdgcn_wmma_f32_16x16x32_f16(false, a0, false, b, (short)0, c01, false, false);
            c11 = __builtin_amdgcn_wmma_f32_16x16x32_f16(false, a1, false, b, (short)0, c11, false, false);
        }
        {
            v8h blo = *(const v8h*)(bp + 2 * bTS);
            v8h bhi = *(const v8h*)(bp + 2 * bTS + 8);
            v16h b;
            #pragma unroll
            for (int j = 0; j < 8; ++j) { b[j] = blo[j]; b[8 + j] = bhi[j]; }
            c02 = __builtin_amdgcn_wmma_f32_16x16x32_f16(false, a0, false, b, (short)0, c02, false, false);
            c12 = __builtin_amdgcn_wmma_f32_16x16x32_f16(false, a1, false, b, (short)0, c12, false, false);
        }
        {
            v8h blo = *(const v8h*)(bp + 3 * bTS);
            v8h bhi = *(const v8h*)(bp + 3 * bTS + 8);
            v16h b;
            #pragma unroll
            for (int j = 0; j < 8; ++j) { b[j] = blo[j]; b[8 + j] = bhi[j]; }
            c03 = __builtin_amdgcn_wmma_f32_16x16x32_f16(false, a0, false, b, (short)0, c03, false, false);
            c13 = __builtin_amdgcn_wmma_f32_16x16x32_f16(false, a1, false, b, (short)0, c13, false, false);
        }
    }

    // Epilogue. C layout: element v -> (m = mtile*16 + v + 8*lhi, n = lane&15)
    const int nlow = lane & 15;
    #pragma unroll
    for (int mt = 0; mt < 2; ++mt) {
        const int mtile = mtile0 + mt;
        #pragma unroll
        for (int t = 0; t < 4; ++t) {
            v8f cc;
            if (mt == 0) cc = (t == 0) ? c00 : (t == 1) ? c01 : (t == 2) ? c02 : c03;
            else         cc = (t == 0) ? c10 : (t == 1) ? c11 : (t == 2) ? c12 : c13;
            int ncol = (ntile0 + t) * 16 + nlow;
            float bv = 0.0f, g = 1.0f, bb = 0.0f;
            if (FLAGS & F_BIAS) bv = bias[ncol];
            if (FLAGS & F_BN) { g = bng[ncol] * BN_INV; bb = bnb[ncol]; }
            #pragma unroll
            for (int v = 0; v < 8; ++v) {
                int mrow = mtile * 16 + v + lhi * 8;
                float y = cc[v] + bv;
                if (FLAGS & F_BN) y = y * g + bb;
                y *= scale;
                long oi = (long)mrow * N + ncol;
                if (FLAGS & F_RES) y += (float)res[oi];
                if (FLAGS & F_RELU) y = y > 0.0f ? y : 0.0f;
                if (FLAGS & F_OUT_H) outh[oi] = (f16)y;
                if (FLAGS & F_OUT_F) outf[oi] = y;
            }
        }
    }
}

// ---------------------------------------------------------------------------
// Depthwise 5x5 "stego" conv, NCHW f32 in -> NHWC f16 out. pad=2, groups=3.
// ---------------------------------------------------------------------------
__global__ void stego_kernel(const float* __restrict__ x, const float* __restrict__ w,
                             const float* __restrict__ b, f16* __restrict__ out)
{
    const int HW = 256 * 256;
    long t = (long)blockIdx.x * blockDim.x + threadIdx.x;
    if (t >= 8L * HW * 3) return;
    int c = (int)(t % 3);
    long r = t / 3;
    int wo = (int)(r % 256); r /= 256;
    int ho = (int)(r % 256);
    int n  = (int)(r / 256);
    float acc = b[c];
    const float* xc = x + ((long)n * 3 + c) * HW;
    const float* wc = w + c * 25;
    #pragma unroll
    for (int ky = 0; ky < 5; ++ky) {
        int hi = ho + ky - 2;
        bool hok = (unsigned)hi < 256u;
        int hcl = hok ? hi : 0;
        #pragma unroll
        for (int kx = 0; kx < 5; ++kx) {
            int wi = wo + kx - 2;
            bool ok = hok && ((unsigned)wi < 256u);
            int wcl = ((unsigned)wi < 256u) ? wi : 0;
            float v = xc[hcl * 256 + wcl];
            acc += (ok ? v : 0.0f) * wc[ky * 5 + kx];
        }
    }
    out[t] = (f16)acc;   // t is exactly the NHWC index
}

// ---------------------------------------------------------------------------
// Sliding-window attention: one wave per (b, head, s). d_head=128 -> 4 f32/lane.
// ---------------------------------------------------------------------------
__global__ __launch_bounds__(128) void swa_kernel(
    const float* __restrict__ q, const float* __restrict__ k,
    const float* __restrict__ v, float* __restrict__ out)
{
    const int S = 1024, D = 512, dh = 128, WA = 8;
    int wave = blockIdx.x * 4 + (threadIdx.x >> 5);
    if (wave >= 8 * 4 * S) return;
    int s = wave % S; int r = wave / S;
    int h = r % 4;    int b = r / 4;
    int lane = threadIdx.x & 31;
    const float* qp = q + ((long)b * S + s) * D + h * dh + lane * 4;
    float q0 = qp[0], q1 = qp[1], q2 = qp[2], q3 = qp[3];

    float sc[17];
    float mx = -3.4e38f;
    #pragma unroll
    for (int wi = 0; wi < 17; ++wi) {
        int idx = s + wi - WA;
        float dot = -3.4e38f;
        if (idx >= 0 && idx < S) {               // wave-uniform
            const float* kp = k + ((long)b * S + idx) * D + h * dh + lane * 4;
            float d0 = q0 * kp[0] + q1 * kp[1] + q2 * kp[2] + q3 * kp[3];
            for (int off = 16; off > 0; off >>= 1) d0 += __shfl_xor(d0, off, 32);
            dot = d0;
        }
        sc[wi] = dot;
        mx = fmaxf(mx, dot);
    }
    float denom = 0.0f;
    #pragma unroll
    for (int wi = 0; wi < 17; ++wi) {
        float e = (sc[wi] > -1e37f) ? expf(sc[wi] - mx) : 0.0f;
        sc[wi] = e; denom += e;
    }
    float inv = 1.0f / denom;
    float o0 = 0, o1 = 0, o2 = 0, o3 = 0;
    #pragma unroll
    for (int wi = 0; wi < 17; ++wi) {
        int idx = s + wi - WA;
        if (idx < 0 || idx >= S) continue;       // wave-uniform
        float p = sc[wi] * inv;
        const float* vp = v + ((long)b * S + idx) * D + h * dh + lane * 4;
        o0 += p * vp[0]; o1 += p * vp[1]; o2 += p * vp[2]; o3 += p * vp[3];
    }
    float* op = out + ((long)b * S + s) * D + h * dh + lane * 4;
    op[0] = o0; op[1] = o1; op[2] = o2; op[3] = o3;
}

// ---------------------------------------------------------------------------
// out = LayerNorm(a + bIn) over D=512; writes f32 (+ optional f16) copies.
// ---------------------------------------------------------------------------
__global__ __launch_bounds__(256) void add_ln_kernel(
    const float* __restrict__ a, const float* __restrict__ bIn,
    const float* __restrict__ g, const float* __restrict__ be,
    float* __restrict__ outf, f16* __restrict__ outh, int rows)
{
    __shared__ float red[16];
    int row = blockIdx.x;
    if (row >= rows) return;
    int tid = threadIdx.x;
    long base = (long)row * 512;
    float x0 = a[base + tid]        + bIn[base + tid];
    float x1 = a[base + 256 + tid]  + bIn[base + 256 + tid];
    float s = x0 + x1;
    float s2 = x0 * x0 + x1 * x1;
    for (int off = 16; off > 0; off >>= 1) {
        s  += __shfl_xor(s, off, 32);
        s2 += __shfl_xor(s2, off, 32);
    }
    int wid = tid >> 5;
    if ((tid & 31) == 0) { red[wid] = s; red[8 + wid] = s2; }
    __syncthreads();
    float S1 = 0, S2 = 0;
    #pragma unroll
    for (int i = 0; i < 8; ++i) { S1 += red[i]; S2 += red[8 + i]; }
    float mu = S1 * (1.0f / 512.0f);
    float var = S2 * (1.0f / 512.0f) - mu * mu;
    float rs = rsqrtf(var + 1e-5f);
    float y0 = (x0 - mu) * rs * g[tid]       + be[tid];
    float y1 = (x1 - mu) * rs * g[256 + tid] + be[256 + tid];
    outf[base + tid] = y0; outf[base + 256 + tid] = y1;
    if (outh) { outh[base + tid] = (f16)y0; outh[base + 256 + tid] = (f16)y1; }
}

__global__ void pool_kernel(const float* __restrict__ seq, float* __restrict__ pooled)
{
    int t = blockIdx.x * blockDim.x + threadIdx.x;   // 8*512
    if (t >= 8 * 512) return;
    int d = t % 512, b = t / 512;
    const float* p = seq + (long)b * 1024 * 512 + d;
    float s = 0.0f;
    for (int i = 0; i < 1024; ++i) s += p[i * 512];
    pooled[t] = s * (1.0f / 1024.0f);
}

__global__ void fc_kernel(const float* __restrict__ pooled, const float* __restrict__ w,
                          float* __restrict__ out)
{
    int t = threadIdx.x;
    if (t >= 16) return;
    int o = t % 2, b = t / 2;
    float s = 0.0f;
    for (int d = 0; d < 512; ++d) s += pooled[b * 512 + d] * w[o * 512 + d];
    out[b * 2 + o] = s;
}

// ---------------------------------------------------------------------------
// Host orchestration. Workspace (212 MiB, lifetime-aliased):
//  [0,64)    MiB bufA (f16 act)   | later: ffn2F(16) seq3F(16) pooled
//  [64,128)  MiB bufB (f16 act)   | later: qF(16) kF(16) vF(16) attnF(16)
//  [128,192) MiB bufC (f16 res)   | later: seq2F(16) seq2H(8) ffn1H(32)
//  [192,196) MiB packed-B scratch
//  [196,212) MiB seqF (f32 b4 output, LN1 residual)
// ---------------------------------------------------------------------------
static inline unsigned cdivu(long a, long b) { return (unsigned)((a + b - 1) / b); }

extern "C" void kernel_launch(void* const* d_in, const int* in_sizes, int n_in,
                              void* d_out, int out_size, void* d_ws, size_t ws_size,
                              hipStream_t stream)
{
    (void)in_sizes; (void)n_in; (void)out_size; (void)ws_size;
    auto F = [&](int i) { return (const float*)d_in[i]; };
    char* ws = (char*)d_ws;
    const size_t MiB = (size_t)1 << 20;

    f16*   bufA   = (f16*)(ws + 0);
    f16*   bufB   = (f16*)(ws + 64 * MiB);
    f16*   bufC   = (f16*)(ws + 128 * MiB);
    f16*   Bp     = (f16*)(ws + 192 * MiB);
    float* seqF   = (float*)(ws + 196 * MiB);
    float* qF     = (float*)(ws + 64 * MiB);
    float* kF     = (float*)(ws + 80 * MiB);
    float* vF     = (float*)(ws + 96 * MiB);
    float* attnF  = (float*)(ws + 112 * MiB);
    float* seq2F  = (float*)(ws + 128 * MiB);
    f16*   seq2H  = (f16*)(ws + 144 * MiB);
    f16*   ffn1H  = (f16*)(ws + 152 * MiB);
    float* ffn2F  = (float*)(ws + 0);
    float* seq3F  = (float*)(ws + 16 * MiB);
    float* pooled = (float*)(ws + 32 * MiB);

    auto packW = [&](const float* W, int N, int K, int Cin, int mode) {
        long tot = (long)(N >> 4) * ((K + 31) >> 5) * 512;
        pack_B_kernel<<<cdivu(tot, 256), 256, 0, stream>>>(W, Bp, N, K, Cin, mode);
    };
    // waves = (M/32) * (N/64); 4 waves per 128-thread block
    auto gblocks = [](long M, long N) { return (unsigned)(((M >> 5) * (N >> 6) + 3) / 4); };

    // stego depthwise 5x5 -> bufA (NHWC f16, 8x256x256x3)
    stego_kernel<<<cdivu(8L * 256 * 256 * 3, 256), 256, 0, stream>>>(F(0), F(1), F(2), bufA);

    // ---- conv trunk: 4 residual blocks ----
    constexpr int FL_SC  = F_BN | F_OUT_H;
    constexpr int FL_C1  = F_BIAS | F_BN | F_RELU | F_OUT_H;
    constexpr int FL_C2  = F_BIAS | F_BN | F_RES | F_OUT_H;
    constexpr int FL_C2L = F_BIAS | F_BN | F_RES | F_OUT_H | F_OUT_F;

    int Hin = 256, Cin = 3;
    const int couts[4] = {64, 128, 256, 512};
    const int strides[4] = {1, 2, 2, 2};
    for (int i = 0; i < 4; ++i) {
        int pb = 3 + 11 * i;
        int Cout = couts[i], st = strides[i];
        int Hout = Hin / st;
        int M = 8 * Hout * Hout;
        int sh1 = 5 + i;                     // log2(Cin) for i>0
        int sh2 = 6 + i;                     // log2(Cout)

        // shortcut: 1x1 conv (stride) + BN -> bufC
        packW(F(pb + 8), Cout, Cin, Cin, 0);
        gemm_wmma<1, FL_SC><<<gblocks(M, Cout), 128, 0, stream>>>(
            bufA, Bp, nullptr, F(pb + 9), F(pb + 10), nullptr, bufC, nullptr,
            M, Cout, Cin, Hout, Hout, Hin, Hin, Cin, st, -1, 1.0f);

        // conv1 3x3 (stride) + bias + BN + ReLU -> bufB
        packW(F(pb + 0), Cout, Cin * 9, Cin, 1);
        if (i == 0)
            gemm_wmma<3, FL_C1><<<gblocks(M, Cout), 128, 0, stream>>>(
                bufA, Bp, F(pb + 1), F(pb + 2), F(pb + 3), nullptr, bufB, nullptr,
                M, Cout, Cin * 9, Hout, Hout, Hin, Hin, Cin, st, -1, 1.0f);
        else
            gemm_wmma<2, FL_C1><<<gblocks(M, Cout), 128, 0, stream>>>(
                bufA, Bp, F(pb + 1), F(pb + 2), F(pb + 3), nullptr, bufB, nullptr,
                M, Cout, Cin * 9, Hout, Hout, Hin, Hin, Cin, st, sh1, 1.0f);

        // conv2 3x3 (stride 1) + bias + BN + shortcut -> bufA (b4 also -> seqF f32)
        packW(F(pb + 4), Cout, Cout * 9, Cout, 1);
        if (i < 3)
            gemm_wmma<2, FL_C2><<<gblocks(M, Cout), 128, 0, stream>>>(
                bufB, Bp, F(pb + 5), F(pb + 6), F(pb + 7), bufC, bufA, nullptr,
                M, Cout, Cout * 9, Hout, Hout, Hout, Hout, Cout, 1, sh2, 1.0f);
        else
            gemm_wmma<2, FL_C2L><<<gblocks(M, Cout), 128, 0, stream>>>(
                bufB, Bp, F(pb + 5), F(pb + 6), F(pb + 7), bufC, bufA, seqF,
                M, Cout, Cout * 9, Hout, Hout, Hout, Hout, Cout, 1, sh2, 1.0f);
        Hin = Hout; Cin = Cout;
    }

    // ---- transformer: bufA = seq f16 [8192, 512]; seqF = f32 copy ----
    constexpr int FL_PROJ = F_BIAS | F_OUT_F;
    constexpr int FL_FFN1 = F_BIAS | F_RELU | F_OUT_H;
    const int Mseq = 8 * 1024, D = 512;
    const float qscale = 0.0883883476483184406f;   // 1/sqrt(128)

    packW(F(47), D, D, D, 0);
    gemm_wmma<0, FL_PROJ><<<gblocks(Mseq, D), 128, 0, stream>>>(
        bufA, Bp, F(48), nullptr, nullptr, nullptr, nullptr, qF,
        Mseq, D, D, 1, 1, 1, 1, 1, 1, -1, qscale);
    packW(F(49), D, D, D, 0);
    gemm_wmma<0, FL_PROJ><<<gblocks(Mseq, D), 128, 0, stream>>>(
        bufA, Bp, F(50), nullptr, nullptr, nullptr, nullptr, kF,
        Mseq, D, D, 1, 1, 1, 1, 1, 1, -1, 1.0f);
    packW(F(51), D, D, D, 0);
    gemm_wmma<0, FL_PROJ><<<gblocks(Mseq, D), 128, 0, stream>>>(
        bufA, Bp, F(52), nullptr, nullptr, nullptr, nullptr, vF,
        Mseq, D, D, 1, 1, 1, 1, 1, 1, -1, 1.0f);

    swa_kernel<<<cdivu(8L * 4 * 1024, 4), 128, 0, stream>>>(qF, kF, vF, attnF);

    add_ln_kernel<<<8192, 256, 0, stream>>>(seqF, attnF, F(57), F(58), seq2F, seq2H, 8192);

    packW(F(53), 2048, D, D, 0);
    gemm_wmma<0, FL_FFN1><<<gblocks(Mseq, 2048), 128, 0, stream>>>(
        seq2H, Bp, F(54), nullptr, nullptr, nullptr, ffn1H, nullptr,
        Mseq, 2048, D, 1, 1, 1, 1, 1, 1, -1, 1.0f);
    packW(F(55), D, 2048, 2048, 0);
    gemm_wmma<0, FL_PROJ><<<gblocks(Mseq, D), 128, 0, stream>>>(
        ffn1H, Bp, F(56), nullptr, nullptr, nullptr, nullptr, ffn2F,
        Mseq, D, 2048, 1, 1, 1, 1, 1, 1, -1, 1.0f);

    add_ln_kernel<<<8192, 256, 0, stream>>>(seq2F, ffn2F, F(59), F(60), seq3F, nullptr, 8192);

    pool_kernel<<<16, 256, 0, stream>>>(seq3F, pooled);
    fc_kernel<<<1, 32, 0, stream>>>(pooled, F(61), (float*)d_out);
}
